// QuantizedLinear_12558484374170
// MI455X (gfx1250) — 8-bit matrix operand order repaired and checked by reading, compiled, not run on hardware
//
#include <hip/hip_runtime.h>
#include <stdint.h>

typedef __attribute__((ext_vector_type(2))) int v2i;
typedef __attribute__((ext_vector_type(4))) int v4i;
typedef __attribute__((ext_vector_type(8))) int v8i;

#define N_ROWS 8192
#define IN_F   4096
#define OUT_F  4096

__global__ __launch_bounds__(256) void pack_x_kernel(const int* __restrict__ x,
                                                     uint32_t* __restrict__ x8,
                                                     const int* __restrict__ izp,
                                                     int count4) {
  int idx = blockIdx.x * 256 + threadIdx.x;
  if (idx >= count4) return;
  const int zp = izp[0];
  int4 v = ((const int4*)x)[idx];
  uint32_t p = ((uint32_t)((v.x - zp) & 0xff))
             | ((uint32_t)((v.y - zp) & 0xff) << 8)
             | ((uint32_t)((v.z - zp) & 0xff) << 16)
             | ((uint32_t)((v.w - zp) & 0xff) << 24);
  x8[idx] = p;
}

__global__ __launch_bounds__(256) void pack_w_kernel(const int* __restrict__ w,
                                                     uint32_t* __restrict__ w8,
                                                     int count4) {
  int idx = blockIdx.x * 256 + threadIdx.x;
  if (idx >= count4) return;
  int4 v = ((const int4*)w)[idx];
  uint32_t p = ((uint32_t)(v.x & 0xff))
             | ((uint32_t)(v.y & 0xff) << 8)
             | ((uint32_t)(v.z & 0xff) << 16)
             | ((uint32_t)(v.w & 0xff) << 24);
  w8[idx] = p;
}

__global__ __launch_bounds__(256) void qgemm_iu8_wmma_kernel(
    const uint8_t* __restrict__ x8,
    const uint8_t* __restrict__ w8,
    const int*     __restrict__ bias,
    const float*   __restrict__ wscale,
    const float*   __restrict__ in_s_p,
    const float*   __restrict__ out_s_p,
    const int*     __restrict__ ozp_p,
    float*         __restrict__ out) {
  const int lane  = threadIdx.x & 31;
  const int wave  = threadIdx.x >> 5;
  const int waveM = wave >> 2;
  const int waveN = wave & 3;

  const int rowBase = blockIdx.y * 128 + waveM * 64;
  const int oBase   = blockIdx.x * 256 + waveN * 64;

  const int mlane = lane & 15;
  const int khalf = lane >> 4;

  const uint8_t* aptr[4];
  const uint8_t* bptr[4];
#pragma unroll
  for (int mi = 0; mi < 4; ++mi)
    aptr[mi] = x8 + (size_t)(rowBase + 16 * mi + mlane) * IN_F + khalf * 8;
#pragma unroll
  for (int ni = 0; ni < 4; ++ni)
    bptr[ni] = w8 + (size_t)(oBase + 16 * ni + mlane) * IN_F + khalf * 8;

  v8i acc[4][4];
#pragma unroll
  for (int mi = 0; mi < 4; ++mi)
#pragma unroll
    for (int ni = 0; ni < 4; ++ni)
      acc[mi][ni] = (v8i){0, 0, 0, 0, 0, 0, 0, 0};

  for (int k0 = 0; k0 < IN_F; k0 += 64) {
    v8i a[4], b[4];
#pragma unroll
    for (int mi = 0; mi < 4; ++mi) {
      const v2i* p = (const v2i*)(aptr[mi] + k0);
      v2i q0 = p[0], q1 = p[2], q2 = p[4], q3 = p[6];
      a[mi] = (v8i){q0.x, q0.y, q1.x, q1.y, q2.x, q2.y, q3.x, q3.y};
    }
#pragma unroll
    for (int ni = 0; ni < 4; ++ni) {
      const v2i* p = (const v2i*)(bptr[ni] + k0);
      v2i r0 = p[0], r1 = p[2], r2 = p[4], r3 = p[6];
      b[ni] = (v8i){r0.x, r0.y, r1.x, r1.y, r2.x, r2.y, r3.x, r3.y};
    }
#pragma unroll
    for (int mi = 0; mi < 4; ++mi)
#pragma unroll
      for (int ni = 0; ni < 4; ++ni)
        acc[mi][ni] = __builtin_amdgcn_wmma_i32_16x16x64_iu8(
             true, a[mi],  true, b[ni],
            acc[mi][ni],  false,  false);
  }

  const float in_s  = in_s_p[0];
  const float out_s = out_s_p[0];
  const float ozp   = (float)ozp_p[0];

#pragma unroll
  for (int ni = 0; ni < 4; ++ni) {
    const int   o  = oBase + 16 * ni + mlane;
    const float sc = in_s * wscale[o] / out_s;
    const float bi = (float)bias[o];
#pragma unroll
    for (int mi = 0; mi < 4; ++mi) {
      const int rbase = rowBase + 16 * mi + khalf * 8;
      float* orow = out + (size_t)rbase * OUT_F + o;
#pragma unroll
      for (int r = 0; r < 8; ++r) {
        float y = ((float)acc[mi][ni][r] + bi) * sc + ozp;
        y = rintf(y);
        y = fminf(127.0f, fmaxf(-128.0f, y));
        orow[(size_t)r * OUT_F] = y;
      }
    }
  }
}

extern "C" void kernel_launch(void* const* d_in, const int* in_sizes, int n_in,
                              void* d_out, int out_size, void* d_ws, size_t ws_size,
                              hipStream_t stream) {
  const int*   x      = (const int*)d_in[0];
  const int*   w      = (const int*)d_in[1];
  const int*   bias   = (const int*)d_in[2];
  const float* wscale = (const float*)d_in[3];
  const float* in_s   = (const float*)d_in[4];
  const float* out_s  = (const float*)d_in[5];
  const int*   izp    = (const int*)d_in[6];
  const int*   ozp    = (const int*)d_in[7];

  uint8_t* x8 = (uint8_t*)d_ws;
  uint8_t* w8 = x8 + (size_t)N_ROWS * IN_F;

  const int cx4 = N_ROWS * IN_F / 4;
  pack_x_kernel<<<cx4 / 256, 256, 0, stream>>>(x, (uint32_t*)x8, izp, cx4);
  const int cw4 = OUT_F * IN_F / 4;
  pack_w_kernel<<<cw4 / 256, 256, 0, stream>>>(w, (uint32_t*)w8, cw4);

  dim3 grid(OUT_F / 256, N_ROWS / 128);
  qgemm_iu8_wmma_kernel<<<grid, 256, 0, stream>>>(
      x8, w8, bias, wscale, in_s, out_s, ozp, (float*)d_out);
}
